// VectorQuantizer_26328149524716
// MI455X (gfx1250) — matrix operand order repaired and checked by reading, not compiled, not run on hardware
//
#include <hip/hip_runtime.h>
#include <hip/hip_bf16.h>

// ---------------------------------------------------------------------------
// VectorQuantizer for MI455X (gfx1250, wave32, WMMA + async-to-LDS staging)
//
//   dist = x @ W^T        [32768 x 4096], K = 256   (bf16 WMMA, f32 accum)
//   ix   = argmin_n dist  (running min fused into the GEMM, dist never stored)
//   out  = normalized Gaussian-weighted sum of W rows in window ix +/- 20
// ---------------------------------------------------------------------------

#define BATCH      32768
#define NUMVEC     4096
#define EMBED      256
#define KNN        20
#define WIN        (2 * KNN + 1)       // 41

// LDS staging geometry: 32 codebook rows (2 WMMA column-tiles) per stage,
// double buffered. Rows padded 256 -> 264 ushorts (528 B = 132 dwords,
// 132 mod 64 = 4) so lane n's ds_load_b128 hits banks 4n..4n+3: conflict-free.
#define ROWS_PER_STAGE 32
#define NSTAGE         (NUMVEC / ROWS_PER_STAGE)   // 128
#define ROWP           264                          // padded row, ushorts
#define BUFU           (ROWS_PER_STAGE * ROWP)      // 8448 ushorts per buffer

typedef __attribute__((ext_vector_type(16))) __bf16 v16bf;
typedef __attribute__((ext_vector_type(8)))  float  v8f;

union ABFrag {                        // 32 bytes: two 16B loads -> one WMMA operand
    uint4 q[2];
    v16bf v;
};

// ---------------------------------------------------------------------------
// f32 -> bf16 (round to nearest even), 2 elements per thread
// ---------------------------------------------------------------------------
__global__ void cvt_f32_bf16(const float* __restrict__ in,
                             unsigned short* __restrict__ out, int n2) {
    int i = blockIdx.x * blockDim.x + threadIdx.x;   // index into pairs
    if (i >= n2) return;
    float2 f = reinterpret_cast<const float2*>(in)[i];
    unsigned int u0 = __float_as_uint(f.x);
    unsigned int u1 = __float_as_uint(f.y);
    u0 = (u0 + 0x7FFFu + ((u0 >> 16) & 1u)) >> 16;
    u1 = (u1 + 0x7FFFu + ((u1 >> 16) & 1u)) >> 16;
    reinterpret_cast<unsigned int*>(out)[i] = (u1 << 16) | (u0 & 0xFFFFu);
}

// ---------------------------------------------------------------------------
// Fused GEMM + argmin with async double-buffered LDS staging of W.
// Block = 256 threads = 8 waves. Wave w owns batch rows [base + 16w, +16).
// A-fragments (x tile, full K=256) stay in VGPRs for the whole kernel.
// Each stage: whole block async-copies 32 W rows (16 KB) into LDS once,
// then every wave runs 2 column-tiles x 8 v_wmma_f32_16x16x32_bf16 from LDS,
// with all 8 B-fragments of a tile preloaded so the ds_load clause flies
// ahead of the WMMA chain (staggered s_wait_dscnt instead of wait-to-0).
//
// Layouts (ISA 7.12.2, wave32, bf16):
//   A 16x32 : lane m+16h holds row m, k = 8h..8h+7 (VGPR0-3) and 16+8h..+7 (VGPR4-7)
//   B 32x16 : lane n+16h holds col n, k = 8h..8h+7 (VGPR0-3) and 16+8h..+7 (VGPR4-7)
//   C 16x16 : lane m+16h, VGPR r -> element (row r+8h, col m)
// ---------------------------------------------------------------------------
__global__ __launch_bounds__(256)
void argmin_gemm(const unsigned short* __restrict__ xb,
                 const unsigned short* __restrict__ Wb,
                 int* __restrict__ ix) {
    __shared__ __attribute__((aligned(16))) unsigned short smem[2 * BUFU];

    const int lane = threadIdx.x & 31;
    const int wave = threadIdx.x >> 5;
    const int m    = lane & 15;        // column-in-tile (B/C), row-in-tile (A)
    const int h    = lane >> 4;        // half-wave selector
    const int rowBase = blockIdx.x * 128 + wave * 16;

    // copy-role decomposition: 32 rows x 512 B per stage, 256 threads,
    // 8 threads per row, 64 B (4 x b128) per thread
    const int crow  = threadIdx.x >> 3;          // 0..31
    const int cpart = threadIdx.x & 7;           // 0..7

    // ---- load A fragments for this wave's 16 batch rows (resident all kernel)
    const unsigned short* xrow = xb + (size_t)(rowBase + m) * EMBED;
    v16bf a[8];
#pragma unroll
    for (int kk = 0; kk < 8; ++kk) {
        ABFrag f;
        f.q[0] = *reinterpret_cast<const uint4*>(xrow + 32 * kk + 8 * h);
        f.q[1] = *reinterpret_cast<const uint4*>(xrow + 32 * kk + 16 + 8 * h);
        a[kk] = f.v;
    }

    float bestD[8];
    int   bestI[8];
#pragma unroll
    for (int r = 0; r < 8; ++r) { bestD[r] = __builtin_inff(); bestI[r] = 0x7FFFFFFF; }

    // async copy of one 32-row stage of W into LDS buffer `buf`
    auto issue_stage = [&](int s, int buf) {
        const unsigned short* gsrc =
            Wb + (size_t)s * (ROWS_PER_STAGE * EMBED) + crow * EMBED + cpart * 32;
        unsigned ldsBase = (unsigned)(uintptr_t)&smem[buf * BUFU + crow * ROWP + cpart * 32];
#pragma unroll
        for (int j = 0; j < 4; ++j) {
            unsigned long long ga = (unsigned long long)(uintptr_t)(gsrc + j * 8);
            unsigned           la = ldsBase + (unsigned)(j * 16);
            asm volatile("global_load_async_to_lds_b128 %0, %1, off"
                         :: "v"(la), "v"(ga) : "memory");
        }
    };

    issue_stage(0, 0);

    for (int s = 0; s < NSTAGE; ++s) {
        const int buf = s & 1;
        // my async copies for stage s have landed in LDS
        asm volatile("s_wait_asynccnt 0x0" ::: "memory");
        // everyone's copies landed AND everyone is done computing stage s-1
        __syncthreads();
        if (s + 1 < NSTAGE) issue_stage(s + 1, buf ^ 1);

#pragma unroll
        for (int tl = 0; tl < 2; ++tl) {
            const int n = s * ROWS_PER_STAGE + tl * 16 + m;   // this lane's column
            const unsigned short* wrow =
                &smem[buf * BUFU + (tl * 16 + m) * ROWP + 8 * h];

            // preload all 8 B fragments of this tile: one clause of 16
            // ds_load_b128, consumed by the WMMA chain with partial waits
            ABFrag f[8];
#pragma unroll
            for (int kk = 0; kk < 8; ++kk) {
                f[kk].q[0] = *reinterpret_cast<const uint4*>(wrow + 32 * kk);
                f[kk].q[1] = *reinterpret_cast<const uint4*>(wrow + 32 * kk + 16);
            }

            v8f c = {};
#pragma unroll
            for (int kk = 0; kk < 8; ++kk) {
                c = __builtin_amdgcn_wmma_f32_16x16x32_bf16(
                        /*neg_a=*/false, a[kk], /*neg_b=*/false, f[kk].v,
                        /*c_mod=*/(short)0, c, /*reuse_a=*/false, /*reuse_b=*/false);
            }
#pragma unroll
            for (int r = 0; r < 8; ++r) {
                float d = c[r];
                if (d < bestD[r]) { bestD[r] = d; bestI[r] = n; }  // ascending n scan
            }
        }
    }

    // ---- argmin reduction across the 16 lanes sharing a half-wave
    //      (rows r+0..7 live in lanes 0-15, rows r+8..15 in lanes 16-31)
#pragma unroll
    for (int r = 0; r < 8; ++r) {
#pragma unroll
        for (int off = 8; off > 0; off >>= 1) {
            float od = __shfl_xor(bestD[r], off, 16);
            int   oi = __shfl_xor(bestI[r], off, 16);
            if (od < bestD[r] || (od == bestD[r] && oi < bestI[r])) {
                bestD[r] = od; bestI[r] = oi;                  // first-occurrence ties
            }
        }
    }
    if (m == 0) {
#pragma unroll
        for (int r = 0; r < 8; ++r)
            ix[rowBase + r + 8 * h] = bestI[r];
    }
}

// ---------------------------------------------------------------------------
// wknn gather: one 64-thread block per sample; thread t owns out[b][4t..4t+3].
// W is 4 MB -> L2 resident; 41 gathered float4 per thread.
// ---------------------------------------------------------------------------
__global__ __launch_bounds__(64)
void wknn_gather(const float* __restrict__ W,
                 const int* __restrict__ ix,
                 float* __restrict__ out) {
    const int b = blockIdx.x;
    const int t = threadIdx.x;
    const int i = ix[b];
    const bool left_edge = (i - KNN) < 0;

    float acc0 = 0.f, acc1 = 0.f, acc2 = 0.f, acc3 = 0.f, wsum = 0.f;
#pragma unroll
    for (int j = 0; j < WIN; ++j) {
        const int off = j - KNN;
        const int idx = i + off;
        const bool valid = (idx >= 0) && (idx < NUMVEC) && (!left_edge || off < KNN);
        if (valid) {
            const float g = expf(-0.5f * (float)(off * off));
            wsum += g;
            const float4 v = *reinterpret_cast<const float4*>(W + (size_t)idx * EMBED + t * 4);
            acc0 += g * v.x; acc1 += g * v.y; acc2 += g * v.z; acc3 += g * v.w;
        }
    }
    const float inv = 1.0f / wsum;
    float4 o = make_float4(acc0 * inv, acc1 * inv, acc2 * inv, acc3 * inv);
    *reinterpret_cast<float4*>(out + (size_t)b * EMBED + t * 4) = o;
}

// ---------------------------------------------------------------------------
extern "C" void kernel_launch(void* const* d_in, const int* in_sizes, int n_in,
                              void* d_out, int out_size, void* d_ws, size_t ws_size,
                              hipStream_t stream) {
    const float* x = (const float*)d_in[0];    // [32768, 256] f32
    const float* W = (const float*)d_in[1];    // [4096, 256]  f32
    float* out = (float*)d_out;                // [32768, 256] f32

    // workspace layout (needs ~19 MB)
    unsigned short* xb = (unsigned short*)d_ws;                       // 16 MB
    unsigned short* Wb = xb + (size_t)BATCH * EMBED;                  //  2 MB
    int*            ix = (int*)(Wb + (size_t)NUMVEC * EMBED);         // 128 KB

    {   // convert x
        int n2 = (BATCH * EMBED) / 2;
        cvt_f32_bf16<<<(n2 + 255) / 256, 256, 0, stream>>>(x, xb, n2);
    }
    {   // convert W
        int n2 = (NUMVEC * EMBED) / 2;
        cvt_f32_bf16<<<(n2 + 255) / 256, 256, 0, stream>>>(W, Wb, n2);
    }

    // fused GEMM + argmin: 256 blocks x 8 waves x 16 rows = 32768 rows
    argmin_gemm<<<BATCH / 128, 256, 0, stream>>>(xb, Wb, ix);

    // neighbor-window weighted gather
    wknn_gather<<<BATCH, 64, 0, stream>>>(W, ix, out);
}
